// TypedBinaryTreeLSTMLayer_50259707298142
// MI455X (gfx1250) — compile-verified, matrix operand order checked by reading, not run on hardware
//
#include <hip/hip_runtime.h>

#define DEV_INLINE __device__ __forceinline__

typedef __attribute__((ext_vector_type(16))) __bf16 v16bf;
typedef __attribute__((ext_vector_type(8)))  __bf16 v8bf;
typedef __attribute__((ext_vector_type(4)))  __bf16 v4bf;
typedef __attribute__((ext_vector_type(8)))  float  v8f;
typedef __attribute__((ext_vector_type(4)))  float  v4f;

#define B_    64
#define L_    256
#define DV_   512
#define T_    32
#define HIN   544      // DV + T
#define KIN   1088     // 2*(DV+T)
#define NGATE 5
#define GOUT  2560     // 5*DV
#define AP    40       // LDS pitch (halfs): 80B rows -> conflict-free 16B frag loads
#define BP    40
#define ASZ   (128 * AP)
#define BSZ   (NGATE * 32 * BP)

DEV_INLINE __bf16 f2bf(float x) {
  unsigned u = __builtin_bit_cast(unsigned, x);
  u += 0x7fffu + ((u >> 16) & 1u);           // round-to-nearest-even
  return __builtin_bit_cast(__bf16, (unsigned short)(u >> 16));
}

DEV_INLINE float sigmoidf_(float x) { return 1.0f / (1.0f + __expf(-x)); }

DEV_INLINE v16bf load_frag(const __bf16* row, int k0) {
  v8bf lo = *(const v8bf*)(row + k0);
  v8bf hi = *(const v8bf*)(row + k0 + 16);
  return __builtin_shufflevector(lo, hi, 0,1,2,3,4,5,6,7,8,9,10,11,12,13,14,15);
}

// ---------------------------------------------------------------------------
// K1: tv = [hl|hr] @ W_comp[tt] (+bias), fused gate math -> c, h_v
// grid (16 col-tiles, 2 row-tiles, 64 batch), 256 threads = 8 waves
// wave = (wrow 0..3)x(wcol 0..1): 32 rows x 16 cols, x5 gates simultaneously.
// LDS ping-pong double buffer + software pipeline: global b128 loads for
// tile ks+1 overlap the 10 WMMAs on tile ks; one barrier per K-step.
// ---------------------------------------------------------------------------
__launch_bounds__(256)
__global__ void comp_gemm_gate_kernel(const float* __restrict__ hl,
                                      const float* __restrict__ hr,
                                      const float* __restrict__ cl,
                                      const float* __restrict__ cr,
                                      const float* __restrict__ W_comp,
                                      const float* __restrict__ b_comp,
                                      const int*  __restrict__ target_types,
                                      float* __restrict__ new_h,
                                      float* __restrict__ out_c) {
  __shared__ __align__(16) __bf16 sA[2 * ASZ];
  __shared__ __align__(16) __bf16 sB[2 * BSZ];

  const int t    = threadIdx.x;
  const int lane = t & 31;
  const int wid  = t >> 5;
  const int wrow = wid >> 1;          // 0..3
  const int wcol = wid & 1;           // 0..1
  const int b    = blockIdx.z;
  const int rowbase = blockIdx.y * 128;
  const int cb   = blockIdx.x * 32;   // gate-space column base
  const int tt   = target_types[b] - 9;
  const float* Wb    = W_comp + (size_t)tt * KIN * GOUT;
  const float* hlrow = hl + ((size_t)b * L_ + rowbase) * HIN;
  const float* hrrow = hr + ((size_t)b * L_ + rowbase) * HIN;

  const int halfsel = lane >> 4;
  const int lrow    = lane & 15;
  const int k0      = halfsel * 8;

  // loop-invariant staging offsets (A: 4 quads/thread, B: 5 quads/thread)
  int aoff_g[4], aoff_l[4];
  #pragma unroll
  for (int i = 0; i < 4; ++i) {
    int qi = i * 256 + t;
    int r = qi >> 3, kq = (qi & 7) * 4;
    aoff_g[i] = r * HIN + kq;
    aoff_l[i] = r * AP + kq;
  }
  int boff_g[5], boff_l[5];
  #pragma unroll
  for (int i = 0; i < 5; ++i) {
    int qi = i * 256 + t;
    int g = qi >> 8, r = qi & 255;
    int k = r >> 3, nq = (r & 7) * 4;
    boff_g[i] = k * GOUT + g * DV_ + cb + nq;
    boff_l[i] = (g * 32 + nq) * BP + k;    // + j*BP for each of the 4 halfs
  }

  v8f acc[NGATE][2];
  #pragma unroll
  for (int g = 0; g < NGATE; ++g)
    #pragma unroll
    for (int rt = 0; rt < 2; ++rt)
      #pragma unroll
      for (int i = 0; i < 8; ++i) acc[g][rt][i] = 0.0f;

  v4f aS[4], bS[5];

  // prologue: stage K-tile 0 into buffer 0
  #pragma unroll
  for (int i = 0; i < 4; ++i) aS[i] = *(const v4f*)(hlrow + aoff_g[i]);
  #pragma unroll
  for (int i = 0; i < 5; ++i) bS[i] = *(const v4f*)(Wb + boff_g[i]);
  #pragma unroll
  for (int i = 0; i < 4; ++i) {
    v4bf p;
    #pragma unroll
    for (int j = 0; j < 4; ++j) p[j] = f2bf(aS[i][j]);
    *(v4bf*)(sA + aoff_l[i]) = p;
  }
  #pragma unroll
  for (int i = 0; i < 5; ++i)
    #pragma unroll
    for (int j = 0; j < 4; ++j) sB[boff_l[i] + j * BP] = f2bf(bS[i][j]);
  __syncthreads();

  for (int ks = 0; ks < KIN / 32; ++ks) {
    const int cur = ks & 1;
    const __bf16* curA = sA + cur * ASZ;
    const __bf16* curB = sB + cur * BSZ;
    const bool more = (ks + 1 < KIN / 32);

    // issue global loads for next K-tile (overlap with WMMA below)
    if (more) {
      const int kg2 = (ks + 1) * 32;
      const float* ap = (kg2 < HIN) ? (hlrow + kg2) : (hrrow + (kg2 - HIN));
      const float* bp = Wb + (size_t)kg2 * GOUT;
      #pragma unroll
      for (int i = 0; i < 4; ++i) aS[i] = *(const v4f*)(ap + aoff_g[i]);
      #pragma unroll
      for (int i = 0; i < 5; ++i) bS[i] = *(const v4f*)(bp + boff_g[i]);
    }

    // preload all fragments, then run the 10 WMMAs back-to-back
    v16bf afrag[2], bfrag[NGATE];
    #pragma unroll
    for (int rt = 0; rt < 2; ++rt)
      afrag[rt] = load_frag(&curA[(wrow * 32 + rt * 16 + lrow) * AP], k0);
    #pragma unroll
    for (int g = 0; g < NGATE; ++g)
      bfrag[g] = load_frag(&curB[(g * 32 + wcol * 16 + lrow) * BP], k0);

    #pragma unroll
    for (int g = 0; g < NGATE; ++g)
      #pragma unroll
      for (int rt = 0; rt < 2; ++rt)
        acc[g][rt] = __builtin_amdgcn_wmma_f32_16x16x32_bf16(
            false, afrag[rt], false, bfrag[g], (short)0, acc[g][rt], false, false);

    // convert + store staged tile into the other buffer
    if (more) {
      __bf16* nxtA = sA + (1 - cur) * ASZ;
      __bf16* nxtB = sB + (1 - cur) * BSZ;
      #pragma unroll
      for (int i = 0; i < 4; ++i) {
        v4bf p;
        #pragma unroll
        for (int j = 0; j < 4; ++j) p[j] = f2bf(aS[i][j]);
        *(v4bf*)(nxtA + aoff_l[i]) = p;
      }
      #pragma unroll
      for (int i = 0; i < 5; ++i)
        #pragma unroll
        for (int j = 0; j < 4; ++j) nxtB[boff_l[i] + j * BP] = f2bf(bS[i][j]);
    }
    __syncthreads();
  }

  // epilogue: bias + LSTM gates, write c and h_v
  const int colg = cb + wcol * 16 + lrow;   // 0..511 within each gate
  float bias[NGATE];
  #pragma unroll
  for (int g = 0; g < NGATE; ++g) bias[g] = b_comp[tt * GOUT + g * DV_ + colg];

  #pragma unroll
  for (int rt = 0; rt < 2; ++rt) {
    #pragma unroll
    for (int v = 0; v < 8; ++v) {
      int l = rowbase + wrow * 32 + rt * 16 + v + halfsel * 8;
      float ig  = acc[0][rt][v] + bias[0];
      float flg = acc[1][rt][v] + bias[1];
      float frg = acc[2][rt][v] + bias[2];
      float ug  = acc[3][rt][v] + bias[3];
      float og  = acc[4][rt][v] + bias[4];
      size_t cidx = ((size_t)b * L_ + l) * DV_ + colg;
      float cc = cl[cidx] * sigmoidf_(flg + 1.0f)
               + cr[cidx] * sigmoidf_(frg + 1.0f)
               + tanhf(ug) * sigmoidf_(ig);
      float hh = sigmoidf_(og) * tanhf(cc);
      out_c[cidx] = cc;
      new_h[((size_t)b * L_ + l) * HIN + colg] = hh;
    }
  }
}

// ---------------------------------------------------------------------------
// K2: logits = h_v @ W_tp + b_tp, softmax -> probs into new_h[...,512:544]
// grid 128 blocks of 128 flat rows (r = b*256+l); 8 waves (4 row x 2 col)
// ---------------------------------------------------------------------------
__launch_bounds__(256)
__global__ void tp_softmax_kernel(const float* __restrict__ W_tp,
                                  const float* __restrict__ b_tp,
                                  float* __restrict__ new_h) {
  __shared__ __align__(16) __bf16 sA[128 * AP];
  __shared__ __align__(16) __bf16 sB[32 * BP];
  __shared__ float sLog[128 * 33];

  const int t    = threadIdx.x;
  const int lane = t & 31;
  const int wid  = t >> 5;
  const int wrow = wid >> 1;
  const int wcol = wid & 1;
  const int rbase = blockIdx.x * 128;
  const int halfsel = lane >> 4;
  const int lrow    = lane & 15;
  const int k0      = halfsel * 8;
  const float* nhrow = new_h + (size_t)rbase * HIN;

  int aoff_g[4], aoff_l[4];
  #pragma unroll
  for (int i = 0; i < 4; ++i) {
    int qi = i * 256 + t;
    int r = qi >> 3, kq = (qi & 7) * 4;
    aoff_g[i] = r * HIN + kq;
    aoff_l[i] = r * AP + kq;
  }
  const int bk = t >> 3, bnq = (t & 7) * 4;   // 1 quad/thread for W_tp

  v8f acc[2];
  #pragma unroll
  for (int rt = 0; rt < 2; ++rt)
    #pragma unroll
    for (int i = 0; i < 8; ++i) acc[rt][i] = 0.0f;

  for (int ks = 0; ks < DV_ / 32; ++ks) {
    const int kg = ks * 32;
    #pragma unroll
    for (int i = 0; i < 4; ++i) {
      v4f a = *(const v4f*)(nhrow + aoff_g[i] + kg);
      v4bf p;
      #pragma unroll
      for (int j = 0; j < 4; ++j) p[j] = f2bf(a[j]);
      *(v4bf*)(sA + aoff_l[i]) = p;
    }
    {
      v4f w = *(const v4f*)(W_tp + (size_t)(kg + bk) * T_ + bnq);
      #pragma unroll
      for (int j = 0; j < 4; ++j) sB[(bnq + j) * BP + bk] = f2bf(w[j]);
    }
    __syncthreads();

    v16bf bfrag = load_frag(&sB[(wcol * 16 + lrow) * BP], k0);
    #pragma unroll
    for (int rt = 0; rt < 2; ++rt) {
      v16bf afrag = load_frag(&sA[(wrow * 32 + rt * 16 + lrow) * AP], k0);
      acc[rt] = __builtin_amdgcn_wmma_f32_16x16x32_bf16(
          false, afrag, false, bfrag, (short)0, acc[rt], false, false);
    }
    __syncthreads();
  }

  const int n = wcol * 16 + lrow;
  const float bias = b_tp[n];
  #pragma unroll
  for (int rt = 0; rt < 2; ++rt)
    #pragma unroll
    for (int v = 0; v < 8; ++v) {
      int m = wrow * 32 + rt * 16 + v + halfsel * 8;
      sLog[m * 33 + n] = acc[rt][v] + bias;
    }
  __syncthreads();

  if (t < 128) {
    float mx = -1e30f;
    #pragma unroll
    for (int j = 0; j < 32; ++j) mx = fmaxf(mx, sLog[t * 33 + j]);
    float e[32]; float s = 0.0f;
    #pragma unroll
    for (int j = 0; j < 32; ++j) { e[j] = __expf(sLog[t * 33 + j] - mx); s += e[j]; }
    float inv = 1.0f / s;
    #pragma unroll
    for (int j = 0; j < 32; ++j)
      new_h[(size_t)(rbase + t) * HIN + DV_ + j] = e[j] * inv;
  }
}

// ---------------------------------------------------------------------------
// K3a: dt_flat = type_emb[tgt] @ W_sem[tt] + b_sem (with target==25 override),
//      broadcast over L into the dt output. grid=B, 64 threads.
// ---------------------------------------------------------------------------
__global__ void sem_dt_kernel(const float* __restrict__ type_emb,
                              const float* __restrict__ W_sem,
                              const float* __restrict__ b_sem,
                              const int*  __restrict__ target_types,
                              float* __restrict__ out_dt) {
  __shared__ float sdt[24];
  const int b = blockIdx.x, t = threadIdx.x;
  const int tgt = target_types[b], tt = tgt - 9;
  if (t < 24) {
    float a = b_sem[tt * 24 + t];
    const float* te = type_emb + (size_t)tgt * DV_;
    const float* Ws = W_sem + (size_t)tt * DV_ * 24 + t;
    for (int d = 0; d < DV_; ++d) a += te[d] * Ws[(size_t)d * 24];
    if (tgt == 25) {              // structural template override
      int k = t / 3, c = t % 3;
      a = (k == 0) ? (c == 0 ? 1.0f : 0.0f) : (c == 2 ? 1.0f : 0.0f);
    }
    sdt[t] = a;
  }
  __syncthreads();
  for (int e = t; e < L_ * 24; e += 64)
    out_dt[(size_t)b * L_ * 24 + e] = sdt[e % 24];
}

// ---------------------------------------------------------------------------
// K3b: loss = -mean_b log_softmax(log(clip(probs[b, pos[b]])))[target[b]]
// ---------------------------------------------------------------------------
__global__ void loss_kernel(const int* __restrict__ target_types,
                            const int* __restrict__ positions,
                            const float* __restrict__ new_h,
                            float* __restrict__ out_loss) {
  __shared__ float red[64];
  const int t = threadIdx.x;           // one thread per batch, 64 threads
  const int pos = positions[t];
  const float* p = new_h + ((size_t)t * L_ + pos) * HIN + DV_;
  float lp[32]; float mx = -1e30f;
  #pragma unroll
  for (int j = 0; j < 32; ++j) {
    float pv = fminf(fmaxf(p[j], 1e-10f), 1.0f);
    lp[j] = logf(pv);
    mx = fmaxf(mx, lp[j]);
  }
  float s = 0.0f;
  #pragma unroll
  for (int j = 0; j < 32; ++j) s += __expf(lp[j] - mx);
  float lsm = lp[target_types[t]] - mx - logf(s);
  red[t] = -lsm * (1.0f / 64.0f);
  __syncthreads();
  for (int off = 32; off > 0; off >>= 1) {
    if (t < off) red[t] += red[t + off];
    __syncthreads();
  }
  if (t == 0) out_loss[0] = red[0];
}

// ---------------------------------------------------------------------------
// K4: template application. One wave per (b,l); lane = output column v.
// ---------------------------------------------------------------------------
__launch_bounds__(256)
__global__ void template_kernel(const float* __restrict__ dl,
                                const float* __restrict__ dr,
                                const float* __restrict__ dt,
                                float* __restrict__ new_d) {
  __shared__ float res[8][16 * 32];
  const int t = threadIdx.x, lane = t & 31, wid = t >> 5;
  const int p = blockIdx.x * 8 + wid;
  const int b = p >> 8, l = p & 255;
  const float* dlp = dl + ((size_t)b * L_ + l) * 512;
  const float* drp = dr + ((size_t)b * L_ + l) * 512;

  // codes[k] = argmax over 3 of dt[b,0,k,:] (wave-uniform; dt broadcast over L)
  const float* dtb = dt + (size_t)b * L_ * 24;
  int codes[8];
  #pragma unroll
  for (int k = 0; k < 8; ++k) {
    float c0 = dtb[k * 3], c1 = dtb[k * 3 + 1], c2 = dtb[k * 3 + 2];
    int cd = 0; float best = c0;
    if (c1 > best) { best = c1; cd = 1; }
    if (c2 > best) { best = c2; cd = 2; }
    codes[k] = cd;
  }

  // len_l / len_r: row's one-hot argmax != 0  <=>  element 0 of the row != 1
  float v0 = (lane < 16) ? dlp[lane * 32] : drp[(lane - 16) * 32];
  unsigned long long bm = __ballot(v0 < 0.5f);
  unsigned mask = (unsigned)bm;
  int len_l = __builtin_popcount(mask & 0xffffu);
  int len_r = __builtin_popcount(mask >> 16);

  #pragma unroll
  for (int m = 0; m < 16; ++m) res[wid][m * 32 + lane] = 0.0f;

  int idx = 0; bool done = false;
  #pragma unroll
  for (int k = 0; k < 8; ++k) {
    int code = codes[k];
    bool c_l = !done && (len_l != 0) && (idx + len_l <= 16) && (code == 0);
    bool c_r = !done && (len_r != 0) && (idx + len_r <= 16) && (code == 1);
    bool c_e = !done && (idx < 16) && (code == 2);
    int ln = c_l ? len_l : (c_r ? len_r : 0);
    const float* srcp = c_l ? dlp : drp;
    for (int m = 0; m < ln; ++m)
      res[wid][(idx + m) * 32 + lane] = srcp[m * 32 + lane];
    if (c_e) { res[wid][idx * 32 + lane] = 1.0f; done = true; }
    idx += ln;
  }

  float* outp = new_d + ((size_t)b * L_ + l) * 512;
  #pragma unroll
  for (int m = 0; m < 16; ++m) outp[m * 32 + lane] = res[wid][m * 32 + lane];
}

// ---------------------------------------------------------------------------
extern "C" void kernel_launch(void* const* d_in, const int* in_sizes, int n_in,
                              void* d_out, int out_size, void* d_ws, size_t ws_size,
                              hipStream_t stream) {
  (void)in_sizes; (void)n_in; (void)out_size; (void)d_ws; (void)ws_size;
  const float* hl        = (const float*)d_in[0];
  const float* hr        = (const float*)d_in[1];
  const float* cl        = (const float*)d_in[2];
  const float* cr        = (const float*)d_in[3];
  const float* dl        = (const float*)d_in[4];
  const float* dr        = (const float*)d_in[5];
  const float* W_comp    = (const float*)d_in[6];
  const float* b_comp    = (const float*)d_in[7];
  const float* W_tp      = (const float*)d_in[8];
  const float* b_tp      = (const float*)d_in[9];
  const float* type_emb  = (const float*)d_in[10];
  const float* W_sem     = (const float*)d_in[11];
  const float* b_sem     = (const float*)d_in[12];
  const int*   tgt       = (const int*)d_in[13];
  const int*   positions = (const int*)d_in[14];

  float* out    = (float*)d_out;
  float* new_h  = out;                                  // (B,L,544)
  float* out_c  = new_h + (size_t)B_ * L_ * HIN;        // (B,L,512)
  float* new_d  = out_c + (size_t)B_ * L_ * DV_;        // (B,L,16,32)
  float* loss   = new_d + (size_t)B_ * L_ * DV_;        // scalar
  float* out_dt = loss + 1;                             // (B,L,8,3)

  comp_gemm_gate_kernel<<<dim3(16, 2, 64), 256, 0, stream>>>(
      hl, hr, cl, cr, W_comp, b_comp, tgt, new_h, out_c);
  tp_softmax_kernel<<<dim3(128), 256, 0, stream>>>(W_tp, b_tp, new_h);
  sem_dt_kernel<<<dim3(64), 64, 0, stream>>>(type_emb, W_sem, b_sem, tgt, out_dt);
  loss_kernel<<<dim3(1), 64, 0, stream>>>(tgt, positions, new_h, loss);
  template_kernel<<<dim3(2048), 256, 0, stream>>>(dl, dr, out_dt, new_d);
}
